// Discriminator_26929444946029
// MI455X (gfx1250) — compile-verified, matrix operand order checked by reading, not run on hardware
//
#include <hip/hip_runtime.h>
#include <hip/hip_bf16.h>

// ---------------------------------------------------------------------------
// GNN discriminator for MI455X (gfx1250, wave32, WMMA).
// All GEMMs use v_wmma_f32_16x16x32_bf16. Features stored bf16; scatter-mean
// accumulates f32 via global_atomic_add_f32 (32-bit voffset + scale_offset
// form). Weights pre-swizzled into WMMA fragment-major order (all fragment
// loads are b128). Per-lane scatter offsets / reciprocal degrees are hoisted
// into registers via b128 LDS loads so the inner scatter loop is add+max+
// atomic only. Two independent WMMA accumulator chains hide hazard NOPs.
// ---------------------------------------------------------------------------

typedef __attribute__((ext_vector_type(16))) __bf16 v16bf;
typedef __attribute__((ext_vector_type(8)))  __bf16 v8bf;
typedef __attribute__((ext_vector_type(8)))  float  v8f;
typedef __attribute__((ext_vector_type(4)))  float  v4f;
typedef __attribute__((ext_vector_type(4)))  int    v4i;

#define NN 50000     // nodes  (50000 = 16 * 3125, exact tiles)
#define EE 800000    // edges  (800000 = 16 * 50000, exact tiles)
#define GG 512       // graphs
#define EF_ 16       // edge feature dim

// ------------------------------ utility kernels ----------------------------

__global__ void zero_f32_kernel(float* __restrict__ p, int n) {
  int i = blockIdx.x * blockDim.x + threadIdx.x;
  if (i < n) p[i] = 0.0f;
}

__global__ void zero_i32_kernel(int* __restrict__ p, int n) {
  int i = blockIdx.x * blockDim.x + threadIdx.x;
  if (i < n) p[i] = 0;
}

__global__ void cvt_bf16_kernel(const float* __restrict__ in,
                                __bf16* __restrict__ out, int n) {
  int i = blockIdx.x * blockDim.x + threadIdx.x;
  if (i < n) out[(unsigned)i] = (__bf16)in[(unsigned)i];
}

// histogram of ids -> cnt (in-degree / per-graph node counts)
__global__ void count_kernel(const int* __restrict__ ids,
                             int* __restrict__ cnt, int n) {
  int i = blockIdx.x * blockDim.x + threadIdx.x;
  if (i < n) atomicAdd(&cnt[ids[i]], 1);
}

// Repack W [Krows x FOUT] (f32, row-major) into bf16 WMMA-B fragment-major:
//   out[(((ct*KCH + kcb)*32 + lane)*16) + t],  lane = {khalf(1b), n(4b)}.
// The 16 values per lane are the exact per-lane VGPR contents of the 32x16
// bf16 B operand (two contiguous 8-elem runs). k >= Krows rows are zero pad.
__global__ void repack_w_kernel(const float* __restrict__ W,
                                __bf16* __restrict__ out,
                                int Krows, int KP, int FOUT) {
  int idx = blockIdx.x * blockDim.x + threadIdx.x;
  if (idx >= KP * FOUT) return;
  int t     = idx & 15;
  int lane  = (idx >> 4) & 31;
  int rest  = idx >> 9;
  int KCH   = KP >> 5;
  int kcb   = rest % KCH;
  int ct    = rest / KCH;
  int n     = lane & 15;
  int khalf = lane >> 4;
  int k   = kcb * 32 + ((t < 8) ? (khalf * 8 + t) : (16 + khalf * 8 + (t - 8)));
  int col = ct * 16 + n;
  out[(unsigned)idx] =
      (k < Krows) ? (__bf16)W[(unsigned)(k * FOUT + col)] : (__bf16)0.0f;
}

// helper: concatenate two v8bf into the v16bf WMMA operand
__device__ __forceinline__ v16bf cat16(v8bf lo, v8bf hi) {
  return __builtin_shufflevector(lo, hi, 0, 1, 2, 3, 4, 5, 6, 7,
                                 8, 9, 10, 11, 12, 13, 14, 15);
}

// ------------------------- edge MLP + scatter-add --------------------------
// m = relu([x[src], e] @ Wm + bm) ; agg[dst] += m   (f32 atomics)
// One block = one 16-edge tile; 4 waves split the FOUT/16 column tiles.
template <int FIN, int FOUT>
__global__ __launch_bounds__(128) void edge_mlp_kernel(
    const __bf16* __restrict__ xin,   // [NN, FIN]  bf16
    const __bf16* __restrict__ ebf,   // [EE, 16]   bf16
    const int*    __restrict__ src,
    const int*    __restrict__ dst,
    const __bf16* __restrict__ Wp,    // fragment-major packed [KP x FOUT]
    const float*  __restrict__ bm,    // [FOUT]
    float*        __restrict__ agg)   // [NN, FOUT] f32, pre-zeroed
{
  constexpr int K   = FIN + EF_;
  constexpr int KP  = (K + 31) & ~31;   // K padded to 32
  constexpr int KCH = KP / 32;          // WMMA K-chunks
  constexpr int CH8 = KP / 8;           // 8-bf16 (16B) staging chunks per row
  constexpr int NT  = FOUT / 16;        // output column tiles

  __shared__ alignas(16) __bf16 As[16][KP];
  __shared__ alignas(16) int srcs[16];
  __shared__ alignas(16) int dstoff[16];  // dst * FOUT (u32-safe, < 12.8M)

  const int tile = blockIdx.x;            // EE/16 tiles, exact
  const int tid  = threadIdx.x;

  if (tid < 16) {
    srcs[tid]   = src[tile * 16 + tid];
    dstoff[tid] = dst[tile * 16 + tid] * FOUT;
  }
  __syncthreads();

  // staged gather: 16B-vector loads, rows = [x[src] | e | 0-pad]
  for (int idx = tid; idx < 16 * CH8; idx += 128) {
    int r = idx / CH8;
    int k = (idx - r * CH8) * 8;
    v8bf v = {};
    if (k + 8 <= FIN)
      v = *(const v8bf*)&xin[(unsigned)(srcs[r] * FIN + k)];
    else if (k + 8 <= K)
      v = *(const v8bf*)&ebf[(unsigned)((tile * 16 + r) * EF_ + (k - FIN))];
    *(v8bf*)&As[r][k] = v;
  }
  __syncthreads();

  const int lane  = tid & 31;
  const int wave  = tid >> 5;
  const int row   = lane & 15;   // A row owned by this lane
  const int khalf = lane >> 4;   // 0: K 0-7/16-23 ; 1: K 8-15/24-31
  const int n     = lane & 15;   // B/D column within tile

  // hoist A fragments (shared across all column tiles)
  v8bf afr[2 * KCH];
#pragma unroll
  for (int kcb = 0; kcb < KCH; ++kcb) {
    afr[2 * kcb]     = *(const v8bf*)&As[row][kcb * 32 + khalf * 8];
    afr[2 * kcb + 1] = *(const v8bf*)&As[row][kcb * 32 + 16 + khalf * 8];
  }

  // hoist the 8 per-lane destination offsets (D rows M=mbase..mbase+7)
  const int mbase = (lane < 16) ? 0 : 8;
  v4i dlo = *(const v4i*)&dstoff[mbase];
  v4i dhi = *(const v4i*)&dstoff[mbase + 4];
  int moff[8];
#pragma unroll
  for (int j = 0; j < 4; ++j) { moff[j] = dlo[j]; moff[4 + j] = dhi[j]; }

  const v8bf* Wp8 = (const v8bf*)Wp;

  auto scatter = [&](const v8f& acc, int col) {
    const float bias = bm[col];
#pragma unroll
    for (int r8 = 0; r8 < 8; ++r8) {
      float v = acc[r8] + bias;
      v = v > 0.0f ? v : 0.0f;                    // relu
      atomicAdd(&agg[(unsigned)(moff[r8] + col)], v);
    }
  };

#pragma unroll
  for (int base = 0; base < NT; base += 8) {
    const int ct0 = base + wave;                  // ct0 < NT (NT >= 4)
    if constexpr (NT >= 8) {
      // two independent accumulator chains: hides WMMA->WMMA hazard NOPs
      const int ct1 = ct0 + 4;
      v8f acc0 = {}, acc1 = {};
#pragma unroll
      for (int kcb = 0; kcb < KCH; ++kcb) {
        v16bf a = cat16(afr[2 * kcb], afr[2 * kcb + 1]);
        unsigned b0 = (unsigned)(((ct0 * KCH + kcb) * 32 + lane) * 2);
        unsigned b1 = (unsigned)(((ct1 * KCH + kcb) * 32 + lane) * 2);
        v16bf B0 = cat16(Wp8[b0], Wp8[b0 + 1]);
        v16bf B1 = cat16(Wp8[b1], Wp8[b1 + 1]);
        acc0 = __builtin_amdgcn_wmma_f32_16x16x32_bf16(
            false, a, false, B0, (short)0, acc0, false, false);
        acc1 = __builtin_amdgcn_wmma_f32_16x16x32_bf16(
            false, a, false, B1, (short)0, acc1, false, false);
      }
      scatter(acc0, ct0 * 16 + n);
      scatter(acc1, ct1 * 16 + n);
    } else {
      v8f acc = {};
#pragma unroll
      for (int kcb = 0; kcb < KCH; ++kcb) {
        v16bf a = cat16(afr[2 * kcb], afr[2 * kcb + 1]);
        unsigned bi = (unsigned)(((ct0 * KCH + kcb) * 32 + lane) * 2);
        v16bf B = cat16(Wp8[bi], Wp8[bi + 1]);
        acc = __builtin_amdgcn_wmma_f32_16x16x32_bf16(
            false, a, false, B, (short)0, acc, false, false);
      }
      scatter(acc, ct0 * 16 + n);
    }
  }
}

// ----------------------------- node update ---------------------------------
// feat_out = relu(agg/deg + x @ Ws + bs), stored bf16 for the next layer.
template <int FIN, int FOUT>
__global__ __launch_bounds__(128) void node_update_kernel(
    const __bf16* __restrict__ xin,   // [NN, FIN] bf16
    const __bf16* __restrict__ Wp,    // fragment-major packed [FIN x FOUT]
    const float*  __restrict__ bs,    // [FOUT]
    const float*  __restrict__ agg,   // [NN, FOUT] f32
    const int*    __restrict__ deg,   // [NN] in-degree
    __bf16*       __restrict__ xout)  // [NN, FOUT] bf16
{
  constexpr int KP  = FIN;            // 32/64/128: already 32-aligned
  constexpr int KCH = KP / 32;
  constexpr int CH8 = KP / 8;
  constexpr int NT  = FOUT / 16;

  __shared__ alignas(16) __bf16 As[16][KP];
  __shared__ alignas(16) float sinv[16];   // 1 / max(deg, 1) per node in tile

  const int tile = blockIdx.x;        // NN/16 tiles, exact
  const int tid  = threadIdx.x;

  if (tid < 16) {
    int d = deg[tile * 16 + tid];
    sinv[tid] = 1.0f / (float)(d > 0 ? d : 1);
  }

  for (int idx = tid; idx < 16 * CH8; idx += 128) {
    int r = idx / CH8;
    int k = (idx - r * CH8) * 8;
    *(v8bf*)&As[r][k] =
        *(const v8bf*)&xin[(unsigned)((tile * 16 + r) * FIN + k)];
  }
  __syncthreads();

  const int lane  = tid & 31;
  const int wave  = tid >> 5;
  const int row   = lane & 15;
  const int khalf = lane >> 4;
  const int n     = lane & 15;

  v8bf afr[2 * KCH];
#pragma unroll
  for (int kcb = 0; kcb < KCH; ++kcb) {
    afr[2 * kcb]     = *(const v8bf*)&As[row][kcb * 32 + khalf * 8];
    afr[2 * kcb + 1] = *(const v8bf*)&As[row][kcb * 32 + 16 + khalf * 8];
  }

  // hoist per-lane reciprocal degrees for D rows M=mbase..mbase+7
  const int mbase = (lane < 16) ? 0 : 8;
  v4f ilo = *(const v4f*)&sinv[mbase];
  v4f ihi = *(const v4f*)&sinv[mbase + 4];
  float minv[8];
#pragma unroll
  for (int j = 0; j < 4; ++j) { minv[j] = ilo[j]; minv[4 + j] = ihi[j]; }
  const int nodeb = tile * 16 + mbase;      // first node of this lane's rows

  const v8bf* Wp8 = (const v8bf*)Wp;

  auto emit = [&](const v8f& acc, int col) {
    const float bias = bs[col];
#pragma unroll
    for (int r8 = 0; r8 < 8; ++r8) {
      const unsigned off = (unsigned)((nodeb + r8) * FOUT + col);
      float v = acc[r8] + agg[off] * minv[r8] + bias;
      v = v > 0.0f ? v : 0.0f;
      xout[off] = (__bf16)v;
    }
  };

#pragma unroll
  for (int base = 0; base < NT; base += 8) {
    const int ct0 = base + wave;
    if constexpr (NT >= 8) {
      const int ct1 = ct0 + 4;
      v8f acc0 = {}, acc1 = {};
#pragma unroll
      for (int kcb = 0; kcb < KCH; ++kcb) {
        v16bf a = cat16(afr[2 * kcb], afr[2 * kcb + 1]);
        unsigned b0 = (unsigned)(((ct0 * KCH + kcb) * 32 + lane) * 2);
        unsigned b1 = (unsigned)(((ct1 * KCH + kcb) * 32 + lane) * 2);
        v16bf B0 = cat16(Wp8[b0], Wp8[b0 + 1]);
        v16bf B1 = cat16(Wp8[b1], Wp8[b1 + 1]);
        acc0 = __builtin_amdgcn_wmma_f32_16x16x32_bf16(
            false, a, false, B0, (short)0, acc0, false, false);
        acc1 = __builtin_amdgcn_wmma_f32_16x16x32_bf16(
            false, a, false, B1, (short)0, acc1, false, false);
      }
      emit(acc0, ct0 * 16 + n);
      emit(acc1, ct1 * 16 + n);
    } else {
      v8f acc = {};
#pragma unroll
      for (int kcb = 0; kcb < KCH; ++kcb) {
        v16bf a = cat16(afr[2 * kcb], afr[2 * kcb + 1]);
        unsigned bi = (unsigned)(((ct0 * KCH + kcb) * 32 + lane) * 2);
        v16bf B = cat16(Wp8[bi], Wp8[bi + 1]);
        acc = __builtin_amdgcn_wmma_f32_16x16x32_bf16(
            false, a, false, B, (short)0, acc, false, false);
      }
      emit(acc, ct0 * 16 + n);
    }
  }
}

// -------------------- classifier + per-graph mean pooling ------------------
template <int FOUT>
__global__ void classify_kernel(const __bf16* __restrict__ feat,
                                const float* __restrict__ Wc,  // [FOUT]
                                const float* __restrict__ bc,  // [1]
                                const int*   __restrict__ gid,
                                float*       __restrict__ gsum) {
  int nid = blockIdx.x * blockDim.x + threadIdx.x;
  if (nid >= NN) return;
  float s = bc[0];
  const v8bf* fr = (const v8bf*)(feat + (unsigned)(nid * FOUT));
#pragma unroll
  for (int c = 0; c < FOUT / 8; ++c) {
    v8bf v = fr[c];
#pragma unroll
    for (int t = 0; t < 8; ++t) s += (float)v[t] * Wc[c * 8 + t];
  }
  atomicAdd(&gsum[gid[nid]], s);
}

__global__ void finalize_kernel(const float* __restrict__ gsum,
                                const int*   __restrict__ gcnt,
                                float*       __restrict__ out) {
  int g = blockIdx.x * blockDim.x + threadIdx.x;
  if (g < GG) {
    int c = gcnt[g];
    out[g] += gsum[g] / (float)(c > 0 ? c : 1);
  }
}

// ------------------------------- host side ---------------------------------

template <int FIN, int FOUT>
static void run_layer(const __bf16* featin, __bf16* featout, const __bf16* ebf,
                      const int* src, const int* dst, const int* gid,
                      const __bf16* Wmp, const float* bm,
                      const __bf16* Wsp, const float* bs,
                      const float* Wc, const float* bc,
                      float* agg, const int* deg,
                      float* gsum, const int* gcnt,
                      float* out, hipStream_t stream) {
  zero_f32_kernel<<<(NN * FOUT + 255) / 256, 256, 0, stream>>>(agg, NN * FOUT);
  edge_mlp_kernel<FIN, FOUT><<<EE / 16, 128, 0, stream>>>(
      featin, ebf, src, dst, Wmp, bm, agg);
  node_update_kernel<FIN, FOUT><<<NN / 16, 128, 0, stream>>>(
      featin, Wsp, bs, agg, deg, featout);
  zero_f32_kernel<<<(GG + 255) / 256, 256, 0, stream>>>(gsum, GG);
  classify_kernel<FOUT><<<(NN + 255) / 256, 256, 0, stream>>>(
      featout, Wc, bc, gid, gsum);
  finalize_kernel<<<(GG + 255) / 256, 256, 0, stream>>>(gsum, gcnt, out);
}

extern "C" void kernel_launch(void* const* d_in, const int* in_sizes, int n_in,
                              void* d_out, int out_size, void* d_ws, size_t ws_size,
                              hipStream_t stream) {
  (void)in_sizes; (void)n_in; (void)out_size; (void)ws_size;

  const float* x   = (const float*)d_in[0];
  const float* ef  = (const float*)d_in[1];
  const int*   src = (const int*)d_in[2];
  const int*   dst = (const int*)d_in[3];
  const int*   gid = (const int*)d_in[4];

  const float *Wm[4], *bm[4], *Ws[4], *bs[4], *Wc[4], *bc[4];
  for (int i = 0; i < 4; ++i) {
    Wm[i] = (const float*)d_in[5 + 6 * i];
    bm[i] = (const float*)d_in[6 + 6 * i];
    Ws[i] = (const float*)d_in[7 + 6 * i];
    bs[i] = (const float*)d_in[8 + 6 * i];
    Wc[i] = (const float*)d_in[9 + 6 * i];
    bc[i] = (const float*)d_in[10 + 6 * i];
  }

  const int fin[4]  = {32, 64, 128, 128};
  const int fout[4] = {64, 128, 128, 256};
  const int kp[4]   = {64, 96, 160, 160};   // (fin+16) padded to 32

  // ---- workspace carve (≈ 128.6 MB) ----
  char* p = (char*)d_ws;
  auto carve = [&](size_t bytes) -> void* {
    void* r = (void*)p;
    p += (bytes + 255) & ~(size_t)255;
    return r;
  };
  __bf16* featA = (__bf16*)carve((size_t)NN * 256 * sizeof(__bf16));
  __bf16* featB = (__bf16*)carve((size_t)NN * 256 * sizeof(__bf16));
  float*  agg   = (float*) carve((size_t)NN * 256 * sizeof(float));
  __bf16* ebf   = (__bf16*)carve((size_t)EE * EF_ * sizeof(__bf16));
  int*    deg   = (int*)   carve((size_t)NN * sizeof(int));
  int*    gcnt  = (int*)   carve((size_t)GG * sizeof(int));
  float*  gsum  = (float*) carve((size_t)GG * sizeof(float));
  __bf16* wmp[4];
  __bf16* wsp[4];
  for (int i = 0; i < 4; ++i) {
    wmp[i] = (__bf16*)carve((size_t)kp[i] * fout[i] * sizeof(__bf16));
    wsp[i] = (__bf16*)carve((size_t)fin[i] * fout[i] * sizeof(__bf16));
  }

  float* out = (float*)d_out;

  // ---- per-launch conversions, weight repack, counts ----
  cvt_bf16_kernel<<<(NN * 32 + 255) / 256, 256, 0, stream>>>(x, featA, NN * 32);
  cvt_bf16_kernel<<<(EE * EF_ + 255) / 256, 256, 0, stream>>>(ef, ebf, EE * EF_);
  for (int i = 0; i < 4; ++i) {
    int nm = kp[i] * fout[i];
    int ns = fin[i] * fout[i];
    repack_w_kernel<<<(nm + 255) / 256, 256, 0, stream>>>(
        Wm[i], wmp[i], fin[i] + EF_, kp[i], fout[i]);
    repack_w_kernel<<<(ns + 255) / 256, 256, 0, stream>>>(
        Ws[i], wsp[i], fin[i], fin[i], fout[i]);
  }
  zero_i32_kernel<<<(NN + 255) / 256, 256, 0, stream>>>(deg, NN);
  count_kernel<<<(EE + 255) / 256, 256, 0, stream>>>(dst, deg, EE);   // in-degree
  zero_i32_kernel<<<(GG + 255) / 256, 256, 0, stream>>>(gcnt, GG);
  count_kernel<<<(NN + 255) / 256, 256, 0, stream>>>(gid, gcnt, NN);  // nodes/graph
  zero_f32_kernel<<<(GG + 255) / 256, 256, 0, stream>>>(out, GG);

  // ---- 4 layers, ping-pong feature buffers ----
  run_layer<32, 64>(featA, featB, ebf, src, dst, gid, wmp[0], bm[0], wsp[0],
                    bs[0], Wc[0], bc[0], agg, deg, gsum, gcnt, out, stream);
  run_layer<64, 128>(featB, featA, ebf, src, dst, gid, wmp[1], bm[1], wsp[1],
                     bs[1], Wc[1], bc[1], agg, deg, gsum, gcnt, out, stream);
  run_layer<128, 128>(featA, featB, ebf, src, dst, gid, wmp[2], bm[2], wsp[2],
                      bs[2], Wc[2], bc[2], agg, deg, gsum, gcnt, out, stream);
  run_layer<128, 256>(featB, featA, ebf, src, dst, gid, wmp[3], bm[3], wsp[3],
                      bs[3], Wc[3], bc[3], agg, deg, gsum, gcnt, out, stream);
}
